// BigGCN_64579128262828
// MI455X (gfx1250) — matrix operand order repaired and checked by reading, not compiled, not run on hardware
//
#include <hip/hip_runtime.h>

#define NN   50000
#define EE   512000
#define HH   128
#define EDD  32
#define LL   10
#define GG   64
#define OUTC 10
#define HEADS 4

typedef __attribute__((ext_vector_type(16))) _Float16 v16h;
typedef __attribute__((ext_vector_type(8)))  float    v8f;

// ---------------------------------------------------------------------------
// small helpers
// ---------------------------------------------------------------------------
static __device__ __forceinline__ void atomicMaxF(float* addr, float val) {
    unsigned int* ua = reinterpret_cast<unsigned int*>(addr);
    unsigned int cur = *ua;
    while (__uint_as_float(cur) < val) {
        unsigned int prev = atomicCAS(ua, cur, __float_as_uint(val));
        if (prev == cur) break;
        cur = prev;
    }
}

__global__ void k_fill(float* __restrict__ p, float v, long long n) {
    long long i = (long long)blockIdx.x * blockDim.x + threadIdx.x;
    long long s = (long long)gridDim.x * blockDim.x;
    for (; i < n; i += s) p[i] = v;
}

__global__ void k_f32_to_f16(const float* __restrict__ in, _Float16* __restrict__ out, long long n) {
    long long i = (long long)blockIdx.x * blockDim.x + threadIdx.x;
    long long s = (long long)gridDim.x * blockDim.x;
    for (; i < n; i += s) out[i] = (_Float16)in[i];
}

// W [nmat][K][Nc] f32 -> Wt [nmat][Nc][K] f16 (transposed for WMMA B-side)
__global__ void k_w_transpose(const float* __restrict__ in, _Float16* __restrict__ out,
                              int nmat, int K, int Nc) {
    long long total = (long long)nmat * K * Nc;
    long long i = (long long)blockIdx.x * blockDim.x + threadIdx.x;
    long long s = (long long)gridDim.x * blockDim.x;
    for (; i < total; i += s) {
        int m   = (int)(i / ((long long)K * Nc));
        int rem = (int)(i % ((long long)K * Nc));
        int k = rem / Nc;
        int n = rem % Nc;
        out[((long long)m * Nc + n) * K + k] = (_Float16)in[i];
    }
}

// ---------------------------------------------------------------------------
// WMMA GEMM: C[M,Nc] = A[M,K](f16) @ Bt[Nc,K](f16)^T + bias, optional relu.
// Block = 256 threads = 8 waves; block tile = 128 rows x 128 cols;
// wave tile = 16 rows x 128 cols (8 accumulators of v8f).
// ---------------------------------------------------------------------------
__global__ __launch_bounds__(256) void k_gemm_wmma(
    const _Float16* __restrict__ A, const _Float16* __restrict__ Bt,
    const float* __restrict__ bias,
    float* __restrict__ Cf, _Float16* __restrict__ Ch,
    int M, int K, int Nc, int relu)
{
    __shared__ _Float16 As[128 * 32];   // 8 KB A slab
    const int tid  = threadIdx.x;
    const int wv   = tid >> 5;
    const int lane = tid & 31;
    const int hlf  = lane >> 4;
    const int l16  = lane & 15;
    const int rowBase = blockIdx.x * 128;
    const int colBase = blockIdx.y * 128;

    v8f acc[8];
#pragma unroll
    for (int i = 0; i < 8; ++i)
#pragma unroll
        for (int j = 0; j < 8; ++j) acc[i][j] = 0.0f;

    for (int kb = 0; kb < K; kb += 32) {
        // cooperative coalesced load of the 128x32 A slab
        {
            int r  = tid >> 1;
            int ko = (tid & 1) * 16;
            int gr = rowBase + r;
            uint4 val = {0u, 0u, 0u, 0u};
            if (gr < M) val = *(const uint4*)(A + (size_t)gr * K + kb + ko);
            *(uint4*)(&As[r * 32 + ko]) = val;
        }
        __syncthreads();

        // A fragment: 16x32 f16, contiguous K (16 per half-wave, +16 K for upper half-wave)
        union { v16h v; unsigned int u[8]; } af;
        const unsigned int* Asu = (const unsigned int*)As;
        int mrow = wv * 16 + l16;
#pragma unroll
        for (int vg = 0; vg < 8; ++vg) {
            int k2 = 2 * vg + 16 * hlf;
            af.u[vg] = Asu[(mrow * 32 + k2) >> 1];
        }

#pragma unroll
        for (int ct = 0; ct < 8; ++ct) {
            int ncol = colBase + ct * 16 + l16;
            const _Float16* bp = Bt + (size_t)ncol * K + kb + 16 * hlf;
            union { v16h v; uint4 q[2]; } bf;
            bf.q[0] = *(const uint4*)(bp);
            bf.q[1] = *(const uint4*)(bp + 8);
            acc[ct] = __builtin_amdgcn_wmma_f32_16x16x32_f16(
                false, af.v, false, bf.v, (short)0, acc[ct], false, false);
        }
        __syncthreads();
    }

#pragma unroll
    for (int ct = 0; ct < 8; ++ct) {
        int ncol = colBase + ct * 16 + l16;
        float bv = bias ? bias[ncol] : 0.0f;
#pragma unroll
        for (int r = 0; r < 8; ++r) {
            int grow = rowBase + wv * 16 + hlf * 8 + r;
            if (grow < M) {
                float v = acc[ct][r] + bv;
                if (relu) v = fmaxf(v, 0.0f);
                if (Cf) Cf[(size_t)grow * Nc + ncol] = v;
                if (Ch) Ch[(size_t)grow * Nc + ncol] = (_Float16)v;
            }
        }
    }
}

// ---------------------------------------------------------------------------
// Fused edge-encoder GEMM + GINE message + scatter-add:
//   agg[dst[e], :] += (edge_attr[e] @ We + be) + hn[src[e], :]
// K = 32 (one slab), Nc = 128. Avoids materializing ea[E,128].
// ---------------------------------------------------------------------------
__global__ __launch_bounds__(256) void k_gemm_edge_fused(
    const _Float16* __restrict__ A,   // [E,32] edge_attr f16
    const _Float16* __restrict__ Bt,  // [128,32] We_l transposed f16
    const float* __restrict__ bias,   // [128]
    const float* __restrict__ hn,     // [N,128]
    const int* __restrict__ srcIdx, const int* __restrict__ dstIdx,
    float* __restrict__ agg,          // [N,128] pre-zeroed
    int M)
{
    __shared__ _Float16 As[128 * 32];
    const int tid  = threadIdx.x;
    const int wv   = tid >> 5;
    const int lane = tid & 31;
    const int hlf  = lane >> 4;
    const int l16  = lane & 15;
    const int rowBase = blockIdx.x * 128;

    v8f acc[8];
#pragma unroll
    for (int i = 0; i < 8; ++i)
#pragma unroll
        for (int j = 0; j < 8; ++j) acc[i][j] = 0.0f;

    {
        int r  = tid >> 1;
        int ko = (tid & 1) * 16;
        int gr = rowBase + r;
        uint4 val = {0u, 0u, 0u, 0u};
        if (gr < M) val = *(const uint4*)(A + (size_t)gr * 32 + ko);
        *(uint4*)(&As[r * 32 + ko]) = val;
    }
    __syncthreads();

    union { v16h v; unsigned int u[8]; } af;
    const unsigned int* Asu = (const unsigned int*)As;
    int mrow = wv * 16 + l16;
#pragma unroll
    for (int vg = 0; vg < 8; ++vg) {
        int k2 = 2 * vg + 16 * hlf;
        af.u[vg] = Asu[(mrow * 32 + k2) >> 1];
    }
#pragma unroll
    for (int ct = 0; ct < 8; ++ct) {
        int ncol = ct * 16 + l16;
        const _Float16* bp = Bt + (size_t)ncol * 32 + 16 * hlf;
        union { v16h v; uint4 q[2]; } bf;
        bf.q[0] = *(const uint4*)(bp);
        bf.q[1] = *(const uint4*)(bp + 8);
        acc[ct] = __builtin_amdgcn_wmma_f32_16x16x32_f16(
            false, af.v, false, bf.v, (short)0, acc[ct], false, false);
    }

#pragma unroll
    for (int r = 0; r < 8; ++r) {
        int e = rowBase + wv * 16 + hlf * 8 + r;
        if (e < M) {
            int s = srcIdx[e];
            int d = dstIdx[e];
#pragma unroll
            for (int ct = 0; ct < 8; ++ct) {
                int ncol = ct * 16 + l16;
                float v = acc[ct][r] + bias[ncol] + hn[(size_t)s * HH + ncol];
                atomicAdd(&agg[(size_t)d * HH + ncol], v);
            }
        }
    }
}

// ---------------------------------------------------------------------------
// BatchNorm (training-mode batch stats over nodes) + ReLU, fused f16 convert
// ---------------------------------------------------------------------------
__global__ void k_bn_stats(const float* __restrict__ x, float* __restrict__ sum,
                           float* __restrict__ sumsq, int nrows) {
    int c  = threadIdx.x & (HH - 1);
    int ro = threadIdx.x >> 7;              // 256 threads -> 2 rows per iter
    float s = 0.0f, s2 = 0.0f;
    for (int r = blockIdx.x * 2 + ro; r < nrows; r += gridDim.x * 2) {
        float v = x[(size_t)r * HH + c];
        s += v; s2 += v * v;
    }
    atomicAdd(&sum[c], s);
    atomicAdd(&sumsq[c], s2);
}

__global__ void k_bn_finalize(const float* __restrict__ sum, const float* __restrict__ sumsq,
                              const float* __restrict__ gamma, const float* __restrict__ beta,
                              float* __restrict__ scale, float* __restrict__ shift) {
    int c = threadIdx.x;
    if (c < HH) {
        float mu  = sum[c] * (1.0f / (float)NN);
        float var = sumsq[c] * (1.0f / (float)NN) - mu * mu;
        float rs  = rsqrtf(var + 1e-5f);
        float sc  = rs * gamma[c];
        scale[c] = sc;
        shift[c] = beta[c] - mu * sc;
    }
}

__global__ void k_bn_apply(const float* __restrict__ o, const float* __restrict__ scale,
                           const float* __restrict__ shift, _Float16* __restrict__ out16,
                           long long n) {
    long long i = (long long)blockIdx.x * blockDim.x + threadIdx.x;
    long long st = (long long)gridDim.x * blockDim.x;
    for (; i < n; i += st) {
        int c = (int)(i & (HH - 1));
        float v = fmaxf(fmaf(o[i], scale[c], shift[c]), 0.0f);
        out16[i] = (_Float16)v;
    }
}

// ---------------------------------------------------------------------------
// GAT kernels
// ---------------------------------------------------------------------------
__global__ void k_gat_scores(const float* __restrict__ h,
                             const float* __restrict__ att_src, const float* __restrict__ att_dst,
                             float* __restrict__ aS, float* __restrict__ aD) {
    int n = blockIdx.x * 8 + (threadIdx.x >> 5);
    if (n >= NN) return;
    int lane = threadIdx.x & 31;
#pragma unroll
    for (int hd = 0; hd < HEADS; ++hd) {
        float ps = 0.0f, pd = 0.0f;
#pragma unroll
        for (int j = 0; j < 4; ++j) {
            int c = lane + j * 32;
            float hv = h[((size_t)n * HEADS + hd) * HH + c];
            ps += hv * att_src[hd * HH + c];
            pd += hv * att_dst[hd * HH + c];
        }
        for (int off = 16; off > 0; off >>= 1) {
            ps += __shfl_down(ps, off, 32);
            pd += __shfl_down(pd, off, 32);
        }
        if (lane == 0) { aS[n * HEADS + hd] = ps; aD[n * HEADS + hd] = pd; }
    }
}

__global__ void k_gat_alpha(const float* __restrict__ aS, const float* __restrict__ aD,
                            const int* __restrict__ srcIdx, const int* __restrict__ dstIdx,
                            float* __restrict__ alphaB, float* __restrict__ amax, int tot) {
    int idx = blockIdx.x * blockDim.x + threadIdx.x;
    if (idx >= tot * HEADS) return;
    int e2 = idx >> 2, hd = idx & 3;
    int s, d;
    if (e2 < EE) { s = srcIdx[e2]; d = dstIdx[e2]; } else { s = d = e2 - EE; }
    float v = aS[s * HEADS + hd] + aD[d * HEADS + hd];
    v = (v > 0.0f) ? v : 0.2f * v;            // leaky relu
    alphaB[idx] = v;
    atomicMaxF(&amax[d * HEADS + hd], v);
}

__global__ void k_gat_expsum(float* __restrict__ alphaB, const float* __restrict__ amax,
                             float* __restrict__ denom, const int* __restrict__ dstIdx, int tot) {
    int idx = blockIdx.x * blockDim.x + threadIdx.x;
    if (idx >= tot * HEADS) return;
    int e2 = idx >> 2, hd = idx & 3;
    int d = (e2 < EE) ? dstIdx[e2] : (e2 - EE);
    float w = __expf(alphaB[idx] - amax[d * HEADS + hd]);
    alphaB[idx] = w;
    atomicAdd(&denom[d * HEADS + hd], w);
}

// head-mean folded in: xg[d,c] += (1/4) * sum_hd coeff_hd * h[s,hd,c]
__global__ void k_gat_agg(const float* __restrict__ h, const float* __restrict__ alphaB,
                          const float* __restrict__ denom,
                          const int* __restrict__ srcIdx, const int* __restrict__ dstIdx,
                          float* __restrict__ xg, int tot) {
    long long n = (long long)tot * 32;
    long long idx = (long long)blockIdx.x * blockDim.x + threadIdx.x;
    long long st = (long long)gridDim.x * blockDim.x;
    for (; idx < n; idx += st) {
        int e2 = (int)(idx >> 5);
        int c  = (int)(idx & 31) * 4;
        int s, d;
        if (e2 < EE) { s = srcIdx[e2]; d = dstIdx[e2]; } else { s = d = e2 - EE; }
        float ax = 0.0f, ay = 0.0f, az = 0.0f, aw = 0.0f;
#pragma unroll
        for (int hd = 0; hd < HEADS; ++hd) {
            float coeff = 0.25f * alphaB[e2 * HEADS + hd] / denom[d * HEADS + hd];
            const float4 hv = *(const float4*)&h[((size_t)s * HEADS + hd) * HH + c];
            ax += coeff * hv.x; ay += coeff * hv.y; az += coeff * hv.z; aw += coeff * hv.w;
        }
        float* dp = &xg[(size_t)d * HH + c];
        atomicAdd(dp + 0, ax); atomicAdd(dp + 1, ay);
        atomicAdd(dp + 2, az); atomicAdd(dp + 3, aw);
    }
}

__global__ void k_pool(const float* __restrict__ xg, const float* __restrict__ bg,
                       const int* __restrict__ batch, float* __restrict__ pooled,
                       float* __restrict__ cnt) {
    long long n = (long long)NN * 32;
    long long idx = (long long)blockIdx.x * blockDim.x + threadIdx.x;
    long long st = (long long)gridDim.x * blockDim.x;
    for (; idx < n; idx += st) {
        int node = (int)(idx >> 5);
        int j    = (int)(idx & 31);
        int c    = j * 4;
        int g    = batch[node];
        const float4 v = *(const float4*)&xg[(size_t)node * HH + c];
        atomicAdd(&pooled[g * HH + c + 0], v.x + bg[c + 0]);
        atomicAdd(&pooled[g * HH + c + 1], v.y + bg[c + 1]);
        atomicAdd(&pooled[g * HH + c + 2], v.z + bg[c + 2]);
        atomicAdd(&pooled[g * HH + c + 3], v.w + bg[c + 3]);
        if (j == 0) atomicAdd(&cnt[g], 1.0f);
    }
}

__global__ void k_fc(const float* __restrict__ pooled, const float* __restrict__ cnt,
                     const float* __restrict__ Wfc, const float* __restrict__ bfc,
                     float* __restrict__ out) {
    int t = blockIdx.x * blockDim.x + threadIdx.x;
    if (t >= GG * OUTC) return;
    int g = t / OUTC, o = t % OUTC;
    float inv = 1.0f / fmaxf(cnt[g], 1.0f);
    float s = bfc[o];
    for (int c = 0; c < HH; ++c) s += pooled[g * HH + c] * inv * Wfc[c * OUTC + o];
    out[t] = s;
}

// ---------------------------------------------------------------------------
// host orchestration
// ---------------------------------------------------------------------------
extern "C" void kernel_launch(void* const* d_in, const int* in_sizes, int n_in,
                              void* d_out, int out_size, void* d_ws, size_t ws_size,
                              hipStream_t stream) {
    (void)in_sizes; (void)n_in; (void)out_size; (void)ws_size;
    const float* x         = (const float*)d_in[0];
    const float* edge_attr = (const float*)d_in[1];
    const int*   eidx      = (const int*)d_in[2];
    const int*   batch     = (const int*)d_in[3];
    const float* Wn   = (const float*)d_in[4];
    const float* bn_b = (const float*)d_in[5];
    const float* We   = (const float*)d_in[6];
    const float* be_b = (const float*)d_in[7];
    const float* W1   = (const float*)d_in[8];
    const float* b1   = (const float*)d_in[9];
    const float* W2   = (const float*)d_in[10];
    const float* b2   = (const float*)d_in[11];
    const float* gamma= (const float*)d_in[12];
    const float* beta = (const float*)d_in[13];
    const float* Wg   = (const float*)d_in[14];
    const float* att_src = (const float*)d_in[15];
    const float* att_dst = (const float*)d_in[16];
    const float* bg   = (const float*)d_in[17];
    const float* Wfc  = (const float*)d_in[18];
    const float* bfc  = (const float*)d_in[19];
    float* out = (float*)d_out;

    const int* srcI = eidx;
    const int* dstI = eidx + EE;

    char* p = (char*)d_ws;
    auto alloc = [&](size_t bytes) -> void* {
        void* r = (void*)p;
        p += (bytes + 255) & ~(size_t)255;
        return r;
    };
    float*    oBuf   = (float*)   alloc((size_t)NN * HH * 4);
    float*    hnBuf  = (float*)   alloc((size_t)NN * HH * 4);
    float*    aggBuf = (float*)   alloc((size_t)NN * HH * 4);        // reused as xg in GAT
    _Float16* actA   = (_Float16*)alloc((size_t)NN * HH * 2);
    _Float16* actB   = (_Float16*)alloc((size_t)NN * HH * 2);
    _Float16* ea16   = (_Float16*)alloc((size_t)EE * EDD * 2);
    float*    hGat   = (float*)   alloc((size_t)NN * HEADS * HH * 4);
    float*    aS     = (float*)   alloc((size_t)NN * HEADS * 4);
    float*    aD     = (float*)   alloc((size_t)NN * HEADS * 4);
    float*    alphaB = (float*)   alloc((size_t)(EE + NN) * HEADS * 4);
    float*    amax   = (float*)   alloc((size_t)NN * HEADS * 4);
    float*    denom  = (float*)   alloc((size_t)NN * HEADS * 4);
    float*    pooled = (float*)   alloc((size_t)GG * HH * 4);
    float*    cnt    = (float*)   alloc((size_t)GG * 4);
    float*    bnsum  = (float*)   alloc(HH * 4);
    float*    bnsq   = (float*)   alloc(HH * 4);
    float*    bnscale= (float*)   alloc(HH * 4);
    float*    bnshift= (float*)   alloc(HH * 4);
    _Float16* Wnt    = (_Float16*)alloc((size_t)LL * HH * HH * 2);
    _Float16* Wet    = (_Float16*)alloc((size_t)LL * HH * EDD * 2);
    _Float16* W1t    = (_Float16*)alloc((size_t)LL * HH * HH * 2);
    _Float16* W2t    = (_Float16*)alloc((size_t)LL * HH * HH * 2);
    _Float16* Wgt    = (_Float16*)alloc((size_t)HEADS * HH * HH * 2);

    // --- preprocess: transpose+convert weights, convert activations ---
    k_w_transpose<<<2048, 256, 0, stream>>>(Wn, Wnt, LL, HH, HH);
    k_w_transpose<<<512,  256, 0, stream>>>(We, Wet, LL, EDD, HH);
    k_w_transpose<<<2048, 256, 0, stream>>>(W1, W1t, LL, HH, HH);
    k_w_transpose<<<2048, 256, 0, stream>>>(W2, W2t, LL, HH, HH);
    k_w_transpose<<<512,  256, 0, stream>>>(Wg, Wgt, 1, HH, HEADS * HH);
    k_f32_to_f16<<<8192, 256, 0, stream>>>(x,         actA, (long long)NN * HH);
    k_f32_to_f16<<<8192, 256, 0, stream>>>(edge_attr, ea16, (long long)EE * EDD);

    dim3 gN((NN + 127) / 128, 1);
    dim3 gE((EE + 127) / 128, 1);

    // --- GINE layers ---
    for (int l = 0; l < LL; ++l) {
        // hn = x @ Wn + bn_b
        k_gemm_wmma<<<gN, 256, 0, stream>>>(actA, Wnt + (size_t)l * HH * HH,
                                            bn_b + l * HH, hnBuf, nullptr, NN, HH, HH, 0);
        // agg = scatter_add(hn[src] + edge_attr@We + be_b)
        k_fill<<<4096, 256, 0, stream>>>(aggBuf, 0.0f, (long long)NN * HH);
        k_gemm_edge_fused<<<gE, 256, 0, stream>>>(ea16, Wet + (size_t)l * HH * EDD,
                                                  be_b + l * HH, hnBuf, srcI, dstI, aggBuf, EE);
        // mlp: relu(agg@W1+b1)@W2+b2
        k_f32_to_f16<<<4096, 256, 0, stream>>>(aggBuf, actB, (long long)NN * HH);
        k_gemm_wmma<<<gN, 256, 0, stream>>>(actB, W1t + (size_t)l * HH * HH,
                                            b1 + l * HH, nullptr, actA, NN, HH, HH, 1);
        k_gemm_wmma<<<gN, 256, 0, stream>>>(actA, W2t + (size_t)l * HH * HH,
                                            b2 + l * HH, oBuf, nullptr, NN, HH, HH, 0);
        // BatchNorm (batch stats) + ReLU, write next-layer f16 input
        k_fill<<<1, 256, 0, stream>>>(bnsum, 0.0f, HH);
        k_fill<<<1, 256, 0, stream>>>(bnsq,  0.0f, HH);
        k_bn_stats<<<512, 256, 0, stream>>>(oBuf, bnsum, bnsq, NN);
        k_bn_finalize<<<1, HH, 0, stream>>>(bnsum, bnsq, gamma + l * HH, beta + l * HH,
                                            bnscale, bnshift);
        k_bn_apply<<<4096, 256, 0, stream>>>(oBuf, bnscale, bnshift, actA, (long long)NN * HH);
    }

    // --- GAT: h = x @ Wg  (Nc = 512 -> grid.y = 4) ---
    dim3 gG((NN + 127) / 128, HEADS);
    k_gemm_wmma<<<gG, 256, 0, stream>>>(actA, Wgt, nullptr, hGat, nullptr,
                                        NN, HH, HEADS * HH, 0);
    k_gat_scores<<<(NN + 7) / 8, 256, 0, stream>>>(hGat, att_src, att_dst, aS, aD);

    const int tot = EE + NN;
    k_fill<<<256, 256, 0, stream>>>(amax,  -1e30f, (long long)NN * HEADS);
    k_fill<<<256, 256, 0, stream>>>(denom,  0.0f,  (long long)NN * HEADS);
    k_gat_alpha<<<(tot * HEADS + 255) / 256, 256, 0, stream>>>(aS, aD, srcI, dstI,
                                                               alphaB, amax, tot);
    k_gat_expsum<<<(tot * HEADS + 255) / 256, 256, 0, stream>>>(alphaB, amax, denom, dstI, tot);
    k_fill<<<4096, 256, 0, stream>>>(aggBuf, 0.0f, (long long)NN * HH);   // xg
    k_gat_agg<<<8192, 256, 0, stream>>>(hGat, alphaB, denom, srcI, dstI, aggBuf, tot);

    // --- global mean pool + fc ---
    k_fill<<<64, 256, 0, stream>>>(pooled, 0.0f, (long long)GG * HH);
    k_fill<<<1, 64, 0, stream>>>(cnt, 0.0f, GG);
    k_pool<<<8192, 256, 0, stream>>>(aggBuf, bg, batch, pooled, cnt);
    k_fc<<<(GG * OUTC + 255) / 256, 256, 0, stream>>>(pooled, cnt, Wfc, bfc, out);
}
